// ExtractLastValidTokenTFLite_8967891714565
// MI455X (gfx1250) — compile-verified, matrix operand order checked by reading, not run on hardware
//
#include <hip/hip_runtime.h>
#include <hip/hip_bf16.h>

// D = A(16x64 u8) * B(64x16 u8) + C ; B = all-ones -> D[m,n] = rowsum(A[m,:])
typedef __attribute__((ext_vector_type(8))) int v8i;

__global__ __launch_bounds__(128)
void extract_last_valid_kernel(const float* __restrict__ dec,
                               const int* __restrict__ mask,
                               float* __restrict__ out,
                               int S, int H) {
    const int b    = blockIdx.x;
    const int tid  = threadIdx.x;
    const int wave = tid >> 5;   // 4 waves of 32
    const int lane = tid & 31;

    __shared__ int s_wavesum[4];
    __shared__ int s_idx;

    // ---- Phase 1: mask sum via V_WMMA_I32_16X16X64_IU8 ----
    // Each wave consumes 1024 mask elements per pass (32 lanes x 32 int32,
    // packed to bytes -> one full 16x64 IU8 A-tile). 4 waves cover S=4096.
    const int* mrow = mask + (size_t)b * (size_t)S;
    int wave_total = 0;

    for (int base = wave * 1024; base < S; base += 4 * 1024) {
        const int4* p = (const int4*)(mrow + base + lane * 32);
        v8i a;
#pragma unroll
        for (int i = 0; i < 8; ++i) {
            int4 m = p[i];                       // global_load_b128, coalesced
            // mask values are 0/1 -> pack 4 int32 into 4 bytes of one dword
            a[i] = m.x | (m.y << 8) | (m.z << 16) | (m.w << 24);
        }
        v8i ones;
#pragma unroll
        for (int i = 0; i < 8; ++i) ones[i] = 0x01010101;
        v8i c = {};
        // (sgn_a, A, sgn_b, B, C, reuse_a, reuse_b) -> v_wmma_i32_16x16x64_iu8
        v8i d = __builtin_amdgcn_wmma_i32_16x16x64_iu8(false, a, false, ones,
                                                       c, false, false);
        // D layout: lane n holds M=0..7 (vgpr0..7) of column n; lane n+16 holds M=8..15.
        int partial = 0;
#pragma unroll
        for (int i = 0; i < 8; ++i) partial += d[i];
        partial += __shfl_xor(partial, 16, 32);   // fold M=0..7 with M=8..15
        wave_total += partial;                    // = #ones in this wave's 1024 elems
    }

    if (lane == 0) s_wavesum[wave] = wave_total;
    __syncthreads();

    if (tid == 0) {
        int tot = s_wavesum[0] + s_wavesum[1] + s_wavesum[2] + s_wavesum[3];
        int idx = tot - 1;
        if (idx < 0) idx = 0;                     // jnp.maximum(sum-1, 0)
        s_idx = idx;
    }
    __syncthreads();

    // ---- Phase 2: gather one H-length row (the whole point: never touch
    // the other S-1 rows, unlike the reference's one-hot multiply-reduce) ----
    const int idx = s_idx;
    const float4* src = (const float4*)(dec + ((size_t)b * (size_t)S + (size_t)idx) * (size_t)H);
    float4*       dst = (float4*)(out + (size_t)b * (size_t)H);
    for (int h4 = tid; h4 < (H >> 2); h4 += blockDim.x)
        dst[h4] = src[h4];                        // b128 load + b128 store
}

extern "C" void kernel_launch(void* const* d_in, const int* in_sizes, int n_in,
                              void* d_out, int out_size, void* d_ws, size_t ws_size,
                              hipStream_t stream) {
    const float* dec  = (const float*)d_in[0];   // [B, S, H] f32
    const int*   mask = (const int*)d_in[1];     // [B, S]    i32
    float*       out  = (float*)d_out;           // [B, H]    f32

    const int BS = in_sizes[1];                  // B*S
    const int H  = in_sizes[0] / BS;             // 512
    const int B  = out_size / H;                 // 32
    const int S  = BS / B;                       // 4096

    dim3 grid(B);
    dim3 block(128);                             // 4 wave32s
    extract_last_valid_kernel<<<grid, block, 0, stream>>>(dec, mask, out, S, H);
}
